// input_timestep_17575006175740
// MI455X (gfx1250) — matrix operand order repaired and checked by reading, not compiled, not run on hardware
//
#include <hip/hip_runtime.h>
#include <hip/hip_bf16.h>

#define DIM         128
#define LDH         136      // padded LDS row stride in halves (272B = 17*16B)
#define BLOCK_ROWS  128      // rows of t handled per block (8 waves x 16 rows)

typedef _Float16 v16h __attribute__((ext_vector_type(16)));
typedef _Float16 v8h  __attribute__((ext_vector_type(8)));
typedef _Float16 v4h  __attribute__((ext_vector_type(4)));
typedef float    v8f  __attribute__((ext_vector_type(8)));
typedef float    v4f  __attribute__((ext_vector_type(4)));

// A operand, 16x32 f16 (ISA 7.12.2): lane L holds row M=L&15,
// K chunks [kg*8, kg*8+8) and [16+kg*8, 16+kg*8+8), kg = L>>4.
__device__ __forceinline__ v16h load_fragA(const _Float16* base, int lane) {
  const int row = lane & 15;
  const int kg  = lane >> 4;
  const _Float16* p = base + row * LDH + kg * 8;
  v8h lo = *(const v8h*)(p);        // K = kg*8 .. +7
  v8h hi = *(const v8h*)(p + 16);   // K = 16+kg*8 .. +7
  v16h r;
#pragma unroll
  for (int i = 0; i < 8; ++i) { r[i] = lo[i]; r[i + 8] = hi[i]; }
  return r;
}

// B operand, 32x16 f16: lane L holds column N=L&15,
// K chunks [kg*8, kg*8+8) and [16+kg*8, 16+kg*8+8), kg = L>>4.  Since we compute
// act @ W^T, column N of B is row (n0+N) of the row-major weight matrix.
__device__ __forceinline__ v16h load_fragB(const _Float16* base, int lane) {
  const int n  = lane & 15;
  const int kg = lane >> 4;
  const _Float16* p = base + n * LDH + kg * 8;
  v8h lo = *(const v8h*)(p);
  v8h hi = *(const v8h*)(p + 16);
  v16h r;
#pragma unroll
  for (int i = 0; i < 8; ++i) { r[i] = lo[i]; r[i + 8] = hi[i]; }
  return r;
}

__device__ __forceinline__ v8f wmma_f16(v16h a, v16h b, v8f c) {
  // (neg_a, A, neg_b, B, c_mod, C, reuse_a, reuse_b)
  return __builtin_amdgcn_wmma_f32_16x16x32_f16(false, a, false, b,
                                                (short)0, c, false, false);
}

__device__ __forceinline__ float silu(float x) {
  return x / (1.0f + __expf(-x));
}

__global__ __launch_bounds__(256)
void ts_mlp_kernel(const float* __restrict__ t,
                   const float* __restrict__ W1, const float* __restrict__ b1,
                   const float* __restrict__ W2, const float* __restrict__ b2,
                   float* __restrict__ out) {
  __shared__ __align__(16) _Float16 sW1[DIM * LDH];
  __shared__ __align__(16) _Float16 sW2[DIM * LDH];
  __shared__ __align__(16) _Float16 sAct[BLOCK_ROWS * LDH];
  __shared__ float sB1[DIM];
  __shared__ float sB2[DIM];

  const int tid  = threadIdx.x;
  const int lane = tid & 31;
  const int wave = tid >> 5;
  const int row0 = blockIdx.x * BLOCK_ROWS;

  // ---- Stage 1: cooperative weight load + f32->f16 convert into LDS ----
  for (int idx = tid; idx < DIM * DIM / 4; idx += 256) {
    const int r  = idx >> 5;          // 32 float4 per row
    const int c4 = (idx & 31) * 4;
    v4f w1 = *(const v4f*)(W1 + r * DIM + c4);
    v4f w2 = *(const v4f*)(W2 + r * DIM + c4);
    v4h h1, h2;
#pragma unroll
    for (int i = 0; i < 4; ++i) { h1[i] = (_Float16)w1[i]; h2[i] = (_Float16)w2[i]; }
    *(v4h*)(sW1 + r * LDH + c4) = h1;
    *(v4h*)(sW2 + r * LDH + c4) = h2;
  }
  if (tid < DIM) { sB1[tid] = b1[tid]; sB2[tid] = b2[tid]; }

  // ---- Stage 2: timestep embedding -> sAct (wave-local rows) ----
  {
    const int   r  = tid >> 1;            // wave w fills rows 16w..16w+15
    const int   c0 = (tid & 1) * 64;
    const float tv = t[row0 + r];
    const float k1 = (3.14159265358979323846f / 128.0f) * tv;           // phase*t
    const float k2 = (3.14159265358979323846f / 128.0f) * (1.0f - tv);  // phase*(1-t)
#pragma unroll 8
    for (int c = 0; c < 64; ++c) {
      const float d = (float)(c0 + c);
      const float e = tv * __cosf(k1 * d) + tv * __sinf(k2 * d);
      sAct[r * LDH + c0 + c] = (_Float16)e;
    }
  }
  __syncthreads();   // only barrier needed: weights visible to all waves

  const int m0 = wave * 16;
  const _Float16* actBase = sAct + m0 * LDH;
  const int N   = lane & 15;
  const int mhi = (lane >> 4) * 8;   // C/D layout: lanes>=16 hold M = v+8

  // ---- Layer 1: x1 = silu(emb @ W1^T + b1), written back f16 in-place ----
  {
    v16h a0 = load_fragA(actBase +  0, lane);
    v16h a1 = load_fragA(actBase + 32, lane);
    v16h a2 = load_fragA(actBase + 64, lane);
    v16h a3 = load_fragA(actBase + 96, lane);
    v8f acc[8];
#pragma unroll
    for (int nt = 0; nt < 8; ++nt) {
      const _Float16* wb = sW1 + nt * 16 * LDH;
      v8f c = {};
      c = wmma_f16(a0, load_fragB(wb +  0, lane), c);
      c = wmma_f16(a1, load_fragB(wb + 32, lane), c);
      c = wmma_f16(a2, load_fragB(wb + 64, lane), c);
      c = wmma_f16(a3, load_fragB(wb + 96, lane), c);
      acc[nt] = c;
    }
#pragma unroll
    for (int nt = 0; nt < 8; ++nt) {
      const float bias = sB1[nt * 16 + N];
#pragma unroll
      for (int v = 0; v < 8; ++v) {
        const float x = silu(acc[nt][v] + bias);
        sAct[(m0 + mhi + v) * LDH + nt * 16 + N] = (_Float16)x;
      }
    }
  }

  // ---- Layer 2: out = silu(x1 @ W2^T + b2), fp32 to global ----
  {
    v16h a0 = load_fragA(actBase +  0, lane);
    v16h a1 = load_fragA(actBase + 32, lane);
    v16h a2 = load_fragA(actBase + 64, lane);
    v16h a3 = load_fragA(actBase + 96, lane);
    v8f acc[8];
#pragma unroll
    for (int nt = 0; nt < 8; ++nt) {
      const _Float16* wb = sW2 + nt * 16 * LDH;
      v8f c = {};
      c = wmma_f16(a0, load_fragB(wb +  0, lane), c);
      c = wmma_f16(a1, load_fragB(wb + 32, lane), c);
      c = wmma_f16(a2, load_fragB(wb + 64, lane), c);
      c = wmma_f16(a3, load_fragB(wb + 96, lane), c);
      acc[nt] = c;
    }
    float* orow = out + (size_t)(row0 + m0 + mhi) * DIM;
#pragma unroll
    for (int nt = 0; nt < 8; ++nt) {
      const float bias = sB2[nt * 16 + N];
#pragma unroll
      for (int v = 0; v < 8; ++v) {
        orow[(size_t)v * DIM + nt * 16 + N] = silu(acc[nt][v] + bias);
      }
    }
  }
}

extern "C" void kernel_launch(void* const* d_in, const int* in_sizes, int n_in,
                              void* d_out, int out_size, void* d_ws, size_t ws_size,
                              hipStream_t stream) {
  const float* t  = (const float*)d_in[0];
  const float* W1 = (const float*)d_in[1];
  const float* b1 = (const float*)d_in[2];
  const float* W2 = (const float*)d_in[3];
  const float* b2 = (const float*)d_in[4];
  float* out = (float*)d_out;
  const int B = in_sizes[0];
  const int grid = B / BLOCK_ROWS;
  ts_mlp_kernel<<<grid, 256, 0, stream>>>(t, W1, b1, W2, b2, out);
}
